// MultiheadAttention_11175504904907
// MI455X (gfx1250) — matrix operand order repaired and checked by reading, not compiled, not run on hardware
//
#include <hip/hip_runtime.h>
#include <hip/hip_bf16.h>
#include <math.h>

// Requires ws_size >= 64 MB (48MB bf16 QKV + 16MB bf16 attention output).

#define D_MODEL   1024
#define NUM_HEADS 16
#define BATCH     4
#define SEQ       2048
#define DK        64
#define BH        (BATCH * NUM_HEADS)
#define LDP       40   // padded LDS row stride (bf16 elems) for 32-wide K panels
#define LDK       72   // padded stride for K-chunk rows (64 + 8)
#define LDV       40   // padded stride for transposed V rows (32 + 8)
#define USE_ASYNC_LDS 1
#define USE_HW_CVT    1   // v_cvt_pk_bf16_f32 via inline asm (gfx1250 VOP3)

typedef __bf16 bf16_t;
typedef __attribute__((ext_vector_type(16))) __bf16 v16bf;
typedef __attribute__((ext_vector_type(8)))  __bf16 v8bf;
typedef __attribute__((ext_vector_type(8)))  float  v8f;

#if USE_HW_CVT
// Packed f32 -> 2 x bf16 (round-to-nearest-even) in one VALU op.
// Clang's builtin is feature-gated to gfx950, but the instruction exists on
// gfx1250; inline asm bypasses the frontend gate.
__device__ __forceinline__ unsigned cvt_pk(float a, float b) {
    unsigned r;
    asm("v_cvt_pk_bf16_f32 %0, %1, %2" : "=v"(r) : "v"(a), "v"(b));
    return r;
}
__device__ __forceinline__ bf16_t cvbf(float f) {
    unsigned p = cvt_pk(f, 0.f);
    unsigned short h = (unsigned short)(p & 0xFFFFu);
    return __builtin_bit_cast(bf16_t, h);
}
__device__ __forceinline__ v8bf pack8(float4 a, float4 b) {
    union { unsigned u[4]; v8bf v; } r;
    r.u[0] = cvt_pk(a.x, a.y);
    r.u[1] = cvt_pk(a.z, a.w);
    r.u[2] = cvt_pk(b.x, b.y);
    r.u[3] = cvt_pk(b.z, b.w);
    return r.v;
}
#else
__device__ __forceinline__ bf16_t cvbf(float f) {
    union { float f; unsigned u; } v; v.f = f;
    unsigned r = v.u + 0x7FFFu + ((v.u >> 16) & 1u);   // round-to-nearest-even
    unsigned short h = (unsigned short)(r >> 16);
    return __builtin_bit_cast(bf16_t, h);
}
__device__ __forceinline__ v8bf pack8(float4 a, float4 b) {
    v8bf r;
    r[0] = cvbf(a.x); r[1] = cvbf(a.y); r[2] = cvbf(a.z); r[3] = cvbf(a.w);
    r[4] = cvbf(b.x); r[5] = cvbf(b.y); r[6] = cvbf(b.z); r[7] = cvbf(b.w);
    return r;
}
#endif

__device__ __forceinline__ v8f wmma_bf16(v16bf a, v16bf b, v8f c) {
    // D = A(16x32 bf16) * B(32x16 bf16) + C(16x16 f32)
    return __builtin_amdgcn_wmma_f32_16x16x32_bf16(
        /*neg_a=*/false, a, /*neg_b=*/false, b,
        /*c_mod=*/(short)0, c, /*reuse_a=*/false, /*reuse_b=*/false);
}

#if USE_ASYNC_LDS
// Async copy of 32 contiguous bytes global -> LDS per lane (two B128 ops).
// INST_OFFSET is added to both the LDS dest and the global source address.
__device__ __forceinline__ void async_copy32(unsigned lds_off, const void* gptr) {
    asm volatile("global_load_async_to_lds_b128 %0, %1, off"
                 :: "v"(lds_off), "v"(gptr) : "memory");
    asm volatile("global_load_async_to_lds_b128 %0, %1, off offset:16"
                 :: "v"(lds_off), "v"(gptr) : "memory");
}
__device__ __forceinline__ void wait_async() {
    asm volatile("s_wait_asynccnt 0x0" ::: "memory");
}
// Low 32 bits of a flat shared address == LDS byte offset (aperture maps
// LDS by discarding the upper bits; see ISA 10.2 per-aperture calc).
__device__ __forceinline__ unsigned lds_off32(const void* p) {
    return (unsigned)(size_t)p;
}
#endif

// A fragment (16x32, MxK) from row-major [m][k] storage.
// Lane L: m=L%16, half=L/16; v0..3 = k[half*8 .. +7], v4..7 = k[16+half*8 .. +7].
__device__ __forceinline__ v16bf load_a_frag(const bf16_t* base, int ld) {
    const int lane = threadIdx.x & 31;
    const int m = lane & 15, half = lane >> 4;
    const bf16_t* p = base + m * ld + half * 8;
    v8bf lo = *(const v8bf*)p;
    v8bf hi = *(const v8bf*)(p + 16);
    v16bf r;
#pragma unroll
    for (int i = 0; i < 8; ++i) { r[i] = lo[i]; r[i + 8] = hi[i]; }
    return r;
}

// B fragment (32x16, KxN) from [n][k] storage (k contiguous per n).
// Lane L: n=L%16, half=L/16; v0..3 = k[half*8 .. +7], v4..7 = k[16+half*8 .. +7].
__device__ __forceinline__ v16bf load_b_frag(const bf16_t* base, int ld) {
    const int lane = threadIdx.x & 31;
    const int n = lane & 15, half = lane >> 4;
    const bf16_t* p = base + n * ld + half * 8;
    v8bf lo = *(const v8bf*)p;
    v8bf hi = *(const v8bf*)(p + 16);
    v16bf r;
#pragma unroll
    for (int i = 0; i < 8; ++i) { r[i] = lo[i]; r[i + 8] = hi[i]; }
    return r;
}

// ---------------------------------------------------------------------------
// Kernel 1: qkv = x @ w_qkv^T, scattered to bf16 Q/K/V [part][B][H][S][64].
// Grid (64, 24), 256 threads (8 waves, 4x2 wave grid, 32Mx64N per wave).
// ---------------------------------------------------------------------------
__global__ __launch_bounds__(256) void qkv_kernel(const float* __restrict__ x,
                                                  const float* __restrict__ w,
                                                  bf16_t* __restrict__ qkvb) {
    __shared__ __align__(16) bf16_t ldsA[128 * LDP];
    __shared__ __align__(16) bf16_t ldsB[128 * LDP];
    const int m0 = blockIdx.x * 128;
    const int n0 = blockIdx.y * 128;
    const int tid = threadIdx.x;
    const int wave = tid >> 5, lane = tid & 31;
    const int mb = (wave & 3) * 32;
    const int nb = (wave >> 2) * 64;
    const int sr = tid >> 1;          // staging row 0..127
    const int sc = (tid & 1) * 16;    // staging col 0 or 16

    v8f acc[2][4] = {};

    for (int k0 = 0; k0 < D_MODEL; k0 += 32) {
        __syncthreads();
        {   // stage A (x, fp32 -> bf16): 128x32
            const float* srcA = x + (size_t)(m0 + sr) * D_MODEL + k0 + sc;
            const float* srcB = w + (size_t)(n0 + sr) * D_MODEL + k0 + sc;
            if (k0 + 32 < D_MODEL) {                 // prefetch next panels
                __builtin_prefetch(srcA + 32, 0, 1);
                __builtin_prefetch(srcB + 32, 0, 1);
            }
            const float4* s4 = (const float4*)srcA;
            float4 f0 = s4[0], f1 = s4[1], f2 = s4[2], f3 = s4[3];
            *(v8bf*)&ldsA[sr * LDP + sc]     = pack8(f0, f1);
            *(v8bf*)&ldsA[sr * LDP + sc + 8] = pack8(f2, f3);
            const float4* w4 = (const float4*)srcB;
            float4 g0 = w4[0], g1 = w4[1], g2 = w4[2], g3 = w4[3];
            *(v8bf*)&ldsB[sr * LDP + sc]     = pack8(g0, g1);
            *(v8bf*)&ldsB[sr * LDP + sc + 8] = pack8(g2, g3);
        }
        __syncthreads();

        v16bf af[2], bfv[4];
        af[0] = load_a_frag(&ldsA[mb * LDP], LDP);
        af[1] = load_a_frag(&ldsA[(mb + 16) * LDP], LDP);
#pragma unroll
        for (int j = 0; j < 4; ++j)
            bfv[j] = load_b_frag(&ldsB[(nb + j * 16) * LDP], LDP);
#pragma unroll
        for (int i = 0; i < 2; ++i)
#pragma unroll
            for (int j = 0; j < 4; ++j)
                acc[i][j] = wmma_bf16(af[i], bfv[j], acc[i][j]);
    }

    // Epilogue: scatter to [part][b][h][s][j] bf16 (part: 0=Q,1=K,2=V).
    const int n = lane & 15, half = lane >> 4;
#pragma unroll
    for (int i = 0; i < 2; ++i) {
#pragma unroll
        for (int j = 0; j < 4; ++j) {
            const int ncol = n0 + nb + j * 16;
            const int part = ncol >> 10;
            const int rem  = ncol & 1023;
            const int h    = rem >> 6;
            const int jb   = (rem & 63) + n;
#pragma unroll
            for (int r = 0; r < 8; ++r) {
                const int gm = m0 + mb + i * 16 + r + half * 8;  // = b*SEQ + s
                const int b  = gm >> 11;
                const int s  = gm & (SEQ - 1);
                const size_t idx =
                    ((((size_t)part * BATCH + b) * NUM_HEADS + h) * SEQ + s) * DK + jb;
                qkvb[idx] = cvbf(acc[i][j][r]);
            }
        }
    }
}

// ---------------------------------------------------------------------------
// Kernel 2: flash attention. Grid = BH * (SEQ/64) blocks, 128 threads
// (4 waves x 16 query rows). 32-key chunks, online softmax.
// ---------------------------------------------------------------------------
__global__ __launch_bounds__(128) void attn_kernel(const bf16_t* __restrict__ Qb,
                                                   const bf16_t* __restrict__ Kb,
                                                   const bf16_t* __restrict__ Vb,
                                                   bf16_t* __restrict__ Ob) {
    __shared__ __align__(16) bf16_t ldsK[32 * LDK];      // [t][j], row major
    __shared__ __align__(16) bf16_t ldsVt[64 * LDV];     // [j][t], transposed
    __shared__ __align__(16) bf16_t ldsP[4][16 * LDP];   // per-wave P (16x32)
    const int bh = blockIdx.x >> 5;          // SEQ/64 == 32 tiles per head
    const int s0 = (blockIdx.x & 31) * 64;
    const int tid = threadIdx.x;
    const int wave = tid >> 5, lane = tid & 31;
    const int n = lane & 15, half = lane >> 4;
    const int vr = tid >> 2;                 // staging row 0..31
    const int vc = (tid & 3) * 16;           // staging col 0/16/32/48

    // Q tile for this wave: 16 rows x 64, kept in registers the whole loop.
    const bf16_t* Qbase = Qb + ((size_t)bh * SEQ + s0 + wave * 16) * DK;
    const v16bf qf0 = load_a_frag(Qbase, DK);       // k 0..31
    const v16bf qf1 = load_a_frag(Qbase + 32, DK);  // k 32..63

    v8f o[4] = {};
    float mrow[8], lrow[8];
#pragma unroll
    for (int r = 0; r < 8; ++r) { mrow[r] = -INFINITY; lrow[r] = 0.f; }

    for (int t0 = 0; t0 < SEQ; t0 += 32) {
        __syncthreads();
        {   // cooperative stage: K chunk row-major (async DMA), V transposed
            const bf16_t* ks = Kb + ((size_t)bh * SEQ + t0 + vr) * DK + vc;
            const bf16_t* vs = Vb + ((size_t)bh * SEQ + t0 + vr) * DK + vc;
            if (t0 + 32 < SEQ) {                     // prefetch next chunks
                __builtin_prefetch(ks + 32 * DK, 0, 1);
                __builtin_prefetch(vs + 32 * DK, 0, 1);
            }
#if USE_ASYNC_LDS
            async_copy32(lds_off32(&ldsK[vr * LDK + vc]), ks);
#else
            *(v8bf*)&ldsK[vr * LDK + vc]     = *(const v8bf*)ks;
            *(v8bf*)&ldsK[vr * LDK + vc + 8] = *(const v8bf*)(ks + 8);
#endif
            v8bf v0 = *(const v8bf*)vs;
            v8bf v1 = *(const v8bf*)(vs + 8);
#pragma unroll
            for (int i = 0; i < 8; ++i) {
                ldsVt[(vc + i) * LDV + vr]     = v0[i];
                ldsVt[(vc + 8 + i) * LDV + vr] = v1[i];
            }
#if USE_ASYNC_LDS
            wait_async();
#endif
        }
        __syncthreads();

        // scores: S[16q x 32t] = Q(16x64) @ K_chunk^T
        v8f sc0 = {}, sc1 = {};
        {
            v16bf kf;
            kf = load_b_frag(&ldsK[0], LDK);            sc0 = wmma_bf16(qf0, kf, sc0);
            kf = load_b_frag(&ldsK[32], LDK);           sc0 = wmma_bf16(qf1, kf, sc0);
            kf = load_b_frag(&ldsK[16 * LDK], LDK);     sc1 = wmma_bf16(qf0, kf, sc1);
            kf = load_b_frag(&ldsK[16 * LDK + 32], LDK);sc1 = wmma_bf16(qf1, kf, sc1);
        }

        // online softmax; rows live at (vgpr r, half), columns across 16 lanes
        bf16_t* pb = ldsP[wave];
#pragma unroll
        for (int r = 0; r < 8; ++r) {
            const float a = sc0[r] * 0.125f;            // 1/sqrt(64)
            const float b = sc1[r] * 0.125f;
            float v = fmaxf(a, b);
#pragma unroll
            for (int d = 8; d >= 1; d >>= 1) v = fmaxf(v, __shfl_xor(v, d, 16));
            const float mn   = fmaxf(mrow[r], v);
            const float corr = __expf(mrow[r] - mn);
            const float p0   = __expf(a - mn);
            const float p1   = __expf(b - mn);
            float sum = p0 + p1;
#pragma unroll
            for (int d = 8; d >= 1; d >>= 1) sum += __shfl_xor(sum, d, 16);
            lrow[r] = lrow[r] * corr + sum;
            mrow[r] = mn;
#pragma unroll
            for (int j = 0; j < 4; ++j) o[j][r] *= corr;
            // C-layout -> row-major P in LDS (same-wave DS ops are in-order)
            pb[(r + half * 8) * LDP + n]      = cvbf(p0);
            pb[(r + half * 8) * LDP + 16 + n] = cvbf(p1);
        }

        // O += P(16x32) @ V_chunk(32x64)
        const v16bf pf = load_a_frag(pb, LDP);
#pragma unroll
        for (int j = 0; j < 4; ++j) {
            v16bf vf = load_b_frag(&ldsVt[(j * 16) * LDV], LDV);
            o[j] = wmma_bf16(pf, vf, o[j]);
        }
    }

    // epilogue: O / l, store bf16 to [b][s][h*64+j] (merged-head layout)
    const int b = bh / NUM_HEADS, h = bh % NUM_HEADS;
#pragma unroll
    for (int j = 0; j < 4; ++j) {
#pragma unroll
        for (int r = 0; r < 8; ++r) {
            const int srow = s0 + wave * 16 + r + half * 8;
            const float val = o[j][r] / lrow[r];
            Ob[((size_t)b * SEQ + srow) * D_MODEL + h * DK + j * 16 + n] = cvbf(val);
        }
    }
}

// ---------------------------------------------------------------------------
// Kernel 3: out = attn(bf16) @ w_o^T -> fp32. Grid (64, 8), 256 threads.
// ---------------------------------------------------------------------------
__global__ __launch_bounds__(256) void oproj_kernel(const bf16_t* __restrict__ A,
                                                    const float* __restrict__ w,
                                                    float* __restrict__ out) {
    __shared__ __align__(16) bf16_t ldsA[128 * LDP];
    __shared__ __align__(16) bf16_t ldsB[128 * LDP];
    const int m0 = blockIdx.x * 128;
    const int n0 = blockIdx.y * 128;
    const int tid = threadIdx.x;
    const int wave = tid >> 5, lane = tid & 31;
    const int mb = (wave & 3) * 32;
    const int nb = (wave >> 2) * 64;
    const int sr = tid >> 1;
    const int sc = (tid & 1) * 16;

    v8f acc[2][4] = {};

    for (int k0 = 0; k0 < D_MODEL; k0 += 32) {
        __syncthreads();
        {   // A already bf16: async DMA straight to LDS
            const bf16_t* src = A + (size_t)(m0 + sr) * D_MODEL + k0 + sc;
#if USE_ASYNC_LDS
            async_copy32(lds_off32(&ldsA[sr * LDP + sc]), src);
#else
            *(v8bf*)&ldsA[sr * LDP + sc]     = *(const v8bf*)src;
            *(v8bf*)&ldsA[sr * LDP + sc + 8] = *(const v8bf*)(src + 8);
#endif
            // w_o fp32 -> bf16
            const float* srcW = w + (size_t)(n0 + sr) * D_MODEL + k0 + sc;
            if (k0 + 32 < D_MODEL) __builtin_prefetch(srcW + 32, 0, 1);
            const float4* w4 = (const float4*)srcW;
            float4 g0 = w4[0], g1 = w4[1], g2 = w4[2], g3 = w4[3];
            *(v8bf*)&ldsB[sr * LDP + sc]     = pack8(g0, g1);
            *(v8bf*)&ldsB[sr * LDP + sc + 8] = pack8(g2, g3);
#if USE_ASYNC_LDS
            wait_async();
#endif
        }
        __syncthreads();

        v16bf af[2], bfv[4];
        af[0] = load_a_frag(&ldsA[mb * LDP], LDP);
        af[1] = load_a_frag(&ldsA[(mb + 16) * LDP], LDP);
#pragma unroll
        for (int j = 0; j < 4; ++j)
            bfv[j] = load_b_frag(&ldsB[(nb + j * 16) * LDP], LDP);
#pragma unroll
        for (int i = 0; i < 2; ++i)
#pragma unroll
            for (int j = 0; j < 4; ++j)
                acc[i][j] = wmma_bf16(af[i], bfv[j], acc[i][j]);
    }

    const int n = lane & 15, half = lane >> 4;
#pragma unroll
    for (int i = 0; i < 2; ++i)
#pragma unroll
        for (int j = 0; j < 4; ++j)
#pragma unroll
            for (int r = 0; r < 8; ++r) {
                const int gm = m0 + mb + i * 16 + r + half * 8;
                out[(size_t)gm * D_MODEL + n0 + nb + j * 16 + n] = acc[i][j][r];
            }
}

// ---------------------------------------------------------------------------
extern "C" void kernel_launch(void* const* d_in, const int* in_sizes, int n_in,
                              void* d_out, int out_size, void* d_ws, size_t ws_size,
                              hipStream_t stream) {
    (void)in_sizes; (void)n_in; (void)out_size; (void)ws_size;
    const float* x     = (const float*)d_in[0];
    const float* w_qkv = (const float*)d_in[1];
    const float* w_o   = (const float*)d_in[2];
    float* out = (float*)d_out;

    bf16_t* qkvb = (bf16_t*)d_ws;
    const size_t perMat = (size_t)BATCH * NUM_HEADS * SEQ * DK;   // 8,388,608 elems
    bf16_t* Qb    = qkvb;
    bf16_t* Kb    = qkvb + perMat;
    bf16_t* Vb    = qkvb + 2 * perMat;
    bf16_t* attnb = qkvb + 3 * perMat;                            // bf16 [B,S,D]

    // 1) QKV projection: M=8192, N=3072, K=1024
    qkv_kernel<<<dim3((BATCH * SEQ) / 128, (3 * D_MODEL) / 128), 256, 0, stream>>>(
        x, w_qkv, qkvb);
    // 2) attention: BH * (SEQ/64) blocks of 128 threads
    attn_kernel<<<dim3(BH * (SEQ / 64)), 128, 0, stream>>>(Qb, Kb, Vb, attnb);
    // 3) output projection: M=8192, N=1024, K=1024
    oproj_kernel<<<dim3((BATCH * SEQ) / 128, D_MODEL / 128), 256, 0, stream>>>(
        attnb, w_o, out);
}
